// Up_57269093925152
// MI455X (gfx1250) — matrix operand order repaired and checked by reading, not compiled, not run on hardware
//
#include <hip/hip_runtime.h>
#include <hip/hip_bf16.h>

// ---------------------------------------------------------------------------
// Problem constants (from reference): B=4, T=6, coarse 32x32, fine 64x64,
// CIN=512, HID=256.
// ---------------------------------------------------------------------------
#define BT        24          // B*T
#define MC        24576       // BT*32*32  coarse rows
#define MF        98304       // BT*64*64  fine rows
#define CIN       512
#define HID       256
#define NUP       1024        // HID*2*2 (o,k,l) columns of up GEMM
#define NL        512         // [self|neigh] concatenated output channels

typedef __attribute__((ext_vector_type(16))) _Float16 v16h;
typedef __attribute__((ext_vector_type(8)))  _Float16 v8h;
typedef __attribute__((ext_vector_type(4)))  _Float16 v4h;
typedef __attribute__((ext_vector_type(2)))  _Float16 v2h;
typedef __attribute__((ext_vector_type(8)))  float    v8f;
typedef __attribute__((ext_vector_type(4)))  float    v4f;
typedef __attribute__((ext_vector_type(2)))  float    v2f;

// ---------------------------------------------------------------------------
// Pack f32 weights (K x N row-major; columns [0,N0) from src0 with stride N0,
// [N0,N) from src1 with stride N-N0) into WMMA B-fragment layout:
//   fragment(kb, nb) = 512 halves; lane L holds 16 contiguous halves:
//   N = nb*16 + (L&15); K = kb*32 + (L>=16 ? 16 : 0) + i   (i = 0..15)
// ---------------------------------------------------------------------------
__global__ void pack_b_kernel(const float* __restrict__ src0,
                              const float* __restrict__ src1,
                              int K, int N, int N0, _Float16* __restrict__ dst) {
    int total = K * N;
    for (int h = blockIdx.x * blockDim.x + threadIdx.x; h < total;
         h += gridDim.x * blockDim.x) {
        int frag = h >> 9;
        int r    = h & 511;
        int lane = r >> 4, i = r & 15;
        int nFr  = N >> 4;
        int kb   = frag / nFr, nb = frag % nFr;
        int n    = nb * 16 + (lane & 15);
        int k    = kb * 32 + ((lane >= 16) ? 16 : 0) + i;
        float v  = (n < N0) ? src0[(size_t)k * N0 + n]
                            : src1[(size_t)k * (N - N0) + (n - N0)];
        dst[h] = (_Float16)v;
    }
}

// x2 (f32, MF x 256) -> xcat[:, 0:256] (f16, row stride 512), 4 elems/thread
__global__ void cvt_x2_kernel(const float* __restrict__ x2,
                              _Float16* __restrict__ xcat) {
    int p = blockIdx.x * blockDim.x + threadIdx.x;   // quad index
    if (p >= (MF * HID) / 4) return;
    int R = p >> 6;               // 64 quads per row of 256
    int c = (p & 63) * 4;
    v4f  s = *reinterpret_cast<const v4f*>(x2 + (size_t)p * 4);
    v4h  d;
    d.x = (_Float16)s.x; d.y = (_Float16)s.y;
    d.z = (_Float16)s.z; d.w = (_Float16)s.w;
    *reinterpret_cast<v4h*>(xcat + (size_t)R * NL + c) = d;
}

// ---------------------------------------------------------------------------
// WMMA GEMM: C[M x N] = A[M x K] * B[K x N]
//   Block: 256 threads = 8 waves. Block tile: 64 (M) x 512 (N).
//   Wave w covers N strip [n0blk + w*64, +64): 4x4 fragment grid -> 16 acc.
//   A staged to LDS in A-fragment layout (K chunks of 256), 16B at a time:
//     frag lane L holds 16 halves: M=(L&15); K order {0..15} (L<16)
//                                           {16..31}              (L>=16)
//   MODE 0: up epilogue  (add up_b, scatter-upsample into xcat ch 256..511)
//   MODE 1: layer epilogue (store f16 y, row-major N=512)
// ---------------------------------------------------------------------------
template <int MODE, bool AF32>
__global__ __launch_bounds__(256)
void gemm_wmma_kernel(const void* __restrict__ Aptr,
                      const _Float16* __restrict__ Bp,
                      const float* __restrict__ bias,
                      _Float16* __restrict__ Out,
                      int K, int N) {
    __shared__ __align__(32) _Float16 As[64 * 256];   // 32 KB

    const int tid   = threadIdx.x;
    const int lane  = tid & 31;
    const int wave  = tid >> 5;
    const int m0    = blockIdx.x * 64;
    const int n0blk = blockIdx.y * 512;
    const int nFr   = N >> 4;
    const int nb0   = (n0blk + wave * 64) >> 4;

    const float*    Af = (const float*)Aptr;
    const _Float16* Ah = (const _Float16*)Aptr;

    v8f acc[4][4];
    const v8f vzero = {0.f, 0.f, 0.f, 0.f, 0.f, 0.f, 0.f, 0.f};
#pragma unroll
    for (int mi = 0; mi < 4; ++mi)
#pragma unroll
        for (int t = 0; t < 4; ++t) acc[mi][t] = vzero;

    for (int kc = 0; kc < K; kc += 256) {
        __syncthreads();
        // Stage 64 rows x 256 K-cols into fragment layout, one 16-byte
        // (8-half) contiguous K-run per iteration: 2048 chunks / block.
#pragma unroll
        for (int cc = 0; cc < 8; ++cc) {
            int c    = cc * 256 + tid;
            int f    = c >> 6;            // fragment id = ks*4 + mi
            int ks   = f >> 2, mi = f & 3;
            int q    = c & 63;
            int ln   = q >> 1;
            int dual = q & 1;             // 0: i=0..7, 1: i=8..15
            int Mi   = mi * 16 + (ln & 15);
            int Kb   = kc + ks * 32 + ((ln >= 16) ? 16 : 0) + dual * 8;
            size_t src = (size_t)(m0 + Mi) * K + Kb;
            v8h d;
            if (AF32) {
                v4f s0 = *reinterpret_cast<const v4f*>(Af + src);
                v4f s1 = *reinterpret_cast<const v4f*>(Af + src + 4);
                d[0] = (_Float16)s0.x; d[1] = (_Float16)s0.y;
                d[2] = (_Float16)s0.z; d[3] = (_Float16)s0.w;
                d[4] = (_Float16)s1.x; d[5] = (_Float16)s1.y;
                d[6] = (_Float16)s1.z; d[7] = (_Float16)s1.w;
            } else {
                d = *reinterpret_cast<const v8h*>(Ah + src);
            }
            *reinterpret_cast<v8h*>(&As[(f << 9) + ln * 16 + dual * 8]) = d;
        }
        __syncthreads();

        for (int ks = 0; ks < 8; ++ks) {
            const int kb = (kc >> 5) + ks;      // global k-step for B frags
            v16h a[4];
#pragma unroll
            for (int mi = 0; mi < 4; ++mi)
                a[mi] = *reinterpret_cast<const v16h*>(
                    &As[((ks * 4 + mi) << 9) + lane * 16]);
#pragma unroll
            for (int t = 0; t < 4; ++t) {
                const v16h b = *reinterpret_cast<const v16h*>(
                    Bp + (((size_t)kb * nFr + nb0 + t) << 9) + lane * 16);
#pragma unroll
                for (int mi = 0; mi < 4; ++mi)
                    acc[mi][t] = __builtin_amdgcn_wmma_f32_16x16x32_f16(
                        false, a[mi], false, b, (short)0, acc[mi][t],
                        false, false);
            }
        }
    }

    // ------------------- epilogue -------------------
    // C layout: element v of acc: row = m0 + mi*16 + v + (lane>=16 ? 8 : 0),
    //           col = n0blk + wave*64 + t*16 + (lane&15)
    const int rofs  = (lane >= 16) ? 8 : 0;
    const int nLane = lane & 15;

    if (MODE == 0) {
        // up scatter: input row r = bt*1024 + i*32 + j ; col = o*4 + k2*2 + l
        // -> xcat[bt*4096 + (2i+k2)*64 + (2j+l)][256 + o]
#pragma unroll
        for (int t = 0; t < 4; ++t) {
            int ncol = n0blk + wave * 64 + t * 16 + nLane;
            int o  = ncol >> 2;
            int k2 = (ncol >> 1) & 1;
            int l  = ncol & 1;
            float bv = bias[o];
#pragma unroll
            for (int mi = 0; mi < 4; ++mi) {
#pragma unroll
                for (int v = 0; v < 8; ++v) {
                    int r    = m0 + mi * 16 + v + rofs;
                    int bt   = r >> 10;
                    int rem  = r & 1023;
                    int ii   = rem >> 5, jj = rem & 31;
                    size_t R = (size_t)bt * 4096 +
                               (size_t)(2 * ii + k2) * 64 + (2 * jj + l);
                    Out[R * NL + HID + o] = (_Float16)(acc[mi][t][v] + bv);
                }
            }
        }
    } else {
#pragma unroll
        for (int t = 0; t < 4; ++t) {
            int ncol = n0blk + wave * 64 + t * 16 + nLane;
#pragma unroll
            for (int mi = 0; mi < 4; ++mi) {
                size_t rowBase = (size_t)(m0 + mi * 16 + rofs) * NL + ncol;
#pragma unroll
                for (int v = 0; v < 8; ++v)
                    Out[rowBase + (size_t)v * NL] = (_Float16)acc[mi][t][v];
            }
        }
    }
}

// ---------------------------------------------------------------------------
// Stencil combine: out = relu(y_self + 0.25 * sum_{4 nbr} y_neigh + b)
// y layout: MF rows x 512 (self 0..255 | neigh 256..511), f16.
// Periodic 4-neighborhood inside each BT tile of 64x64. 2 channels/thread.
// ---------------------------------------------------------------------------
template <bool F32OUT>
__global__ void combine_kernel(const _Float16* __restrict__ y,
                               const float* __restrict__ b,
                               void* __restrict__ outp) {
    int p = blockIdx.x * blockDim.x + threadIdx.x;   // pair index
    if (p >= (MF * HID) / 2) return;
    int R  = p >> 7;              // 128 pairs per row of 256
    int d  = (p & 127) * 2;
    int bt = R >> 12;
    int rem = R & 4095;
    int hh  = rem >> 6, ww = rem & 63;
    size_t base = (size_t)bt * 4096;
    size_t Ru = base + (size_t)((hh + 63) & 63) * 64 + ww;
    size_t Rd = base + (size_t)((hh + 1) & 63) * 64 + ww;
    size_t Rl = base + (size_t)hh * 64 + ((ww + 63) & 63);
    size_t Rr = base + (size_t)hh * 64 + ((ww + 1) & 63);

    v2h s  = *reinterpret_cast<const v2h*>(y + (size_t)R * NL + d);
    v2h n0 = *reinterpret_cast<const v2h*>(y + Ru * NL + HID + d);
    v2h n1 = *reinterpret_cast<const v2h*>(y + Rd * NL + HID + d);
    v2h n2 = *reinterpret_cast<const v2h*>(y + Rl * NL + HID + d);
    v2h n3 = *reinterpret_cast<const v2h*>(y + Rr * NL + HID + d);
    v2f bb = *reinterpret_cast<const v2f*>(b + d);

    float v0 = (float)s.x + 0.25f * ((float)n0.x + (float)n1.x +
                                     (float)n2.x + (float)n3.x) + bb.x;
    float v1 = (float)s.y + 0.25f * ((float)n0.y + (float)n1.y +
                                     (float)n2.y + (float)n3.y) + bb.y;
    v0 = v0 > 0.f ? v0 : 0.f;
    v1 = v1 > 0.f ? v1 : 0.f;
    if (F32OUT) {
        v2f o; o.x = v0; o.y = v1;
        *reinterpret_cast<v2f*>((float*)outp + (size_t)p * 2) = o;
    } else {
        v2h o; o.x = (_Float16)v0; o.y = (_Float16)v1;
        *reinterpret_cast<v2h*>((_Float16*)outp + (size_t)p * 2) = o;
    }
}

// ---------------------------------------------------------------------------
extern "C" void kernel_launch(void* const* d_in, const int* in_sizes, int n_in,
                              void* d_out, int out_size, void* d_ws,
                              size_t ws_size, hipStream_t stream) {
    const float* x1      = (const float*)d_in[0];  // (BT,32,32,512)
    const float* x2      = (const float*)d_in[1];  // (BT,64,64,256)
    const float* up_w    = (const float*)d_in[2];  // (512,256,2,2) == K x 1024
    const float* up_b    = (const float*)d_in[3];  // (256)
    const float* w_self1 = (const float*)d_in[4];  // (512,256)
    const float* w_nei1  = (const float*)d_in[5];
    const float* b1      = (const float*)d_in[6];
    const float* w_self2 = (const float*)d_in[7];  // (256,256)
    const float* w_nei2  = (const float*)d_in[8];
    const float* b2      = (const float*)d_in[9];
    float* out = (float*)d_out;

    // workspace layout (f16 buffers), 256B aligned
    char* ws = (char*)d_ws;
    size_t off = 0;
    auto take = [&](size_t bytes) {
        size_t o = off;
        off = (off + bytes + 255) & ~(size_t)255;
        return o;
    };
    _Float16* Bup  = (_Float16*)(ws + take((size_t)CIN * NUP * 2)); // 1 MB
    _Float16* B1   = (_Float16*)(ws + take((size_t)CIN * NL * 2));  // 512 KB
    _Float16* B2   = (_Float16*)(ws + take((size_t)HID * NL * 2));  // 256 KB
    _Float16* xcat = (_Float16*)(ws + take((size_t)MF * NL * 2));   // 96 MB
    _Float16* y    = (_Float16*)(ws + take((size_t)MF * NL * 2));   // 96 MB (y1 & y2)
    _Float16* h1   = (_Float16*)(ws + take((size_t)MF * HID * 2));  // 48 MB
    (void)ws_size;

    // 1) pack weights into WMMA B-fragment layout
    pack_b_kernel<<<(CIN * NUP + 255) / 256, 256, 0, stream>>>(
        up_w, up_w, CIN, NUP, NUP, Bup);
    pack_b_kernel<<<(CIN * NL + 255) / 256, 256, 0, stream>>>(
        w_self1, w_nei1, CIN, NL, HID, B1);
    pack_b_kernel<<<(HID * NL + 255) / 256, 256, 0, stream>>>(
        w_self2, w_nei2, HID, NL, HID, B2);

    // 2) x2 -> xcat[:, 0:256]
    cvt_x2_kernel<<<(MF * HID / 4 + 255) / 256, 256, 0, stream>>>(x2, xcat);

    // 3) Up GEMM (f32 A) -> scatter into xcat[:, 256:512] with bias
    gemm_wmma_kernel<0, true><<<dim3(MC / 64, NUP / 512), 256, 0, stream>>>(
        (const void*)x1, Bup, up_b, xcat, CIN, NUP);

    // 4) SAGE layer 1 GEMM: y = xcat @ [Wself1 | Wneigh1]
    gemm_wmma_kernel<1, false><<<dim3(MF / 64, 1), 256, 0, stream>>>(
        (const void*)xcat, B1, nullptr, y, CIN, NL);

    // 5) combine 1 -> h1 (f16)
    combine_kernel<false><<<(MF * HID / 2 + 255) / 256, 256, 0, stream>>>(
        y, b1, (void*)h1);

    // 6) SAGE layer 2 GEMM: y = h1 @ [Wself2 | Wneigh2]
    gemm_wmma_kernel<1, false><<<dim3(MF / 64, 1), 256, 0, stream>>>(
        (const void*)h1, B2, nullptr, y, HID, NL);

    // 7) combine 2 -> out (f32)
    combine_kernel<true><<<(MF * HID / 2 + 255) / 256, 256, 0, stream>>>(
        y, b2, (void*)out);
}
